// GATLayer_68384469286917
// MI455X (gfx1250) — matrix operand order repaired and checked by reading, not compiled, not run on hardware
//
#include <hip/hip_runtime.h>
#include <hip/hip_bf16.h>

typedef __attribute__((ext_vector_type(16))) __bf16 v16bf;
typedef __attribute__((ext_vector_type(8)))  float  v8f;
typedef __attribute__((ext_vector_type(4)))  unsigned int v4u;
typedef __attribute__((ext_vector_type(8)))  int  v8i;
typedef __attribute__((ext_vector_type(4)))  int  v4i;

#define N_NODES 8192
#define D_MODEL 256
#define N_HEAD  4
#define D_HEAD  64

// ---------- helpers ----------
// native bf16 convert (gfx1250 has hardware bf16 cvt; avoids SW RNE VALU bloat)
__device__ __forceinline__ __bf16 f2bf(float f) { return (__bf16)f; }
__device__ __forceinline__ unsigned short f2bfbits(float f) {
  __bf16 b = (__bf16)f;
  return __builtin_bit_cast(unsigned short, b);
}
__device__ __forceinline__ __bf16 bits2bf(unsigned short s) {
  return __builtin_bit_cast(__bf16, s);
}
// ISA 7.12.2: 16-bit A 16x32 — lane holds row M=lane%16; element j -> K
__device__ __forceinline__ int amap_k(int j, int lane) {
  return ((j < 8) ? j : j + 8) + ((lane < 16) ? 0 : 8);
}
// ISA 7.12.2: 16-bit B 32x16 — lane holds col N=lane%16; element j -> K
__device__ __forceinline__ int bmap_k(int j, int lane) {
  return ((j < 8) ? j : j + 8) + ((lane < 16) ? 0 : 8);
}
// reductions across a 16-lane half-wave (wave32)
__device__ __forceinline__ float half16_max(float v) {
#pragma unroll
  for (int off = 1; off < 16; off <<= 1) v = fmaxf(v, __shfl_xor(v, off, 32));
  return v;
}
__device__ __forceinline__ float half16_sum(float v) {
#pragma unroll
  for (int off = 1; off < 16; off <<= 1) v += __shfl_xor(v, off, 32);
  return v;
}

// ---------- Tensor Data Mover: build D# and issue tensor_load_to_lds ----------
// 2D tile load (tile_d1 rows of tile_d0 elements, row stride stride0 elements)
// data_size_code: 0=1B,1=2B,2=4B,3=8B  (ISA 8.4 group1 bits 17:16)
__device__ __forceinline__ void tdm_load_2d(unsigned lds_addr, const void* gptr,
                                            unsigned data_size_code,
                                            unsigned tile_d0, unsigned tile_d1,
                                            unsigned long long stride0) {
  const unsigned long long ga = (unsigned long long)gptr;
  const unsigned long long td0 = 1u << 20;  // generous tensor dims: no OOB clip
  const unsigned long long td1 = 1u << 20;
  v4u g0;
  g0[0] = 1u;                                   // count=1, user descriptor
  g0[1] = lds_addr;                             // lds_addr (bytes)
  g0[2] = (unsigned)(ga & 0xFFFFFFFFu);         // global_addr[31:0]
  g0[3] = (unsigned)((ga >> 32) & 0x01FFFFFFu)  // global_addr[56:32]
          | (2u << 30);                         // type = 2 ("image")
  v8i g1;
  g1[0] = (int)(data_size_code << 16);          // wg_mask=0 | data_size
  g1[1] = (int)((td0 & 0xFFFFu) << 16);         // tensor_dim0[15:0]
  g1[2] = (int)(((td0 >> 16) & 0xFFFFu) | ((td1 & 0xFFFFu) << 16));
  g1[3] = (int)(((td1 >> 16) & 0xFFFFu) | ((tile_d0 & 0xFFFFu) << 16));
  g1[4] = (int)(tile_d1 & 0xFFFFu);             // tile_dim1 | tile_dim2=0
  g1[5] = (int)(stride0 & 0xFFFFFFFFu);         // dim0_stride[31:0]
  g1[6] = (int)((stride0 >> 32) & 0xFFFFu);     // dim0_stride[47:32] | d1s[15:0]=0
  g1[7] = 0;
  const v4i z4 = {0, 0, 0, 0};                  // 2D tensor: groups 2/3 unused
  const v8i z8 = {0, 0, 0, 0, 0, 0, 0, 0};      // 6-arg (clang-23) extra group
  __builtin_amdgcn_tensor_load_to_lds(g0, g1, z4, z4, z8, 0);
}
__device__ __forceinline__ unsigned lds_off(const void* p) {
  return (unsigned)(size_t)p;                   // generic LDS addr low 32 = offset
}

// ---------- kernel 1: Q/K/V projections (bf16 WMMA, f32 accumulate) ----------
__global__ __launch_bounds__(32) void qkv_proj_kernel(
    const float* __restrict__ x,
    const float* __restrict__ Wq, const float* __restrict__ bq,
    const float* __restrict__ Wk, const float* __restrict__ bk,
    const float* __restrict__ Wv, const float* __restrict__ bv,
    unsigned short* __restrict__ qb, unsigned short* __restrict__ kb,
    unsigned short* __restrict__ vb) {
  const int lane = threadIdx.x & 31;
  const int l15  = lane & 15;
  const int row0 = blockIdx.x * 16;
  const int col0 = blockIdx.y * 16;
  const int wsel = blockIdx.z;

  const float* W    = (wsel == 0) ? Wq : ((wsel == 1) ? Wk : Wv);
  const float* bias = (wsel == 0) ? bq : ((wsel == 1) ? bk : bv);
  unsigned short* ob = (wsel == 0) ? qb : ((wsel == 1) ? kb : vb);

  v8f acc = {0.f, 0.f, 0.f, 0.f, 0.f, 0.f, 0.f, 0.f};
#pragma unroll
  for (int kk = 0; kk < D_MODEL; kk += 32) {
    v16bf a, b;
#pragma unroll
    for (int j = 0; j < 16; ++j)
      a[j] = f2bf(x[(size_t)(row0 + l15) * D_MODEL + kk + amap_k(j, lane)]);
#pragma unroll
    for (int j = 0; j < 16; ++j)
      b[j] = f2bf(W[(size_t)(kk + bmap_k(j, lane)) * D_MODEL + col0 + l15]);
    acc = __builtin_amdgcn_wmma_f32_16x16x32_bf16(false, a, false, b,
                                                  (short)0, acc, false, false);
  }
  const float bcol = bias[col0 + l15];
  const int col = col0 + l15;
  const int h = col >> 6, dk = col & 63;
#pragma unroll
  for (int r = 0; r < 8; ++r) {
    const int row = row0 + r + ((lane < 16) ? 0 : 8);
    ob[((size_t)h * N_NODES + row) * D_HEAD + dk] = f2bfbits(acc[r] + bcol);
  }
}

// ---------- kernel 2: flash-style masked attention with TDM staging ----------
__global__ __launch_bounds__(128) void gat_attn_kernel(
    const unsigned short* __restrict__ qb,
    const unsigned short* __restrict__ kb,
    const unsigned short* __restrict__ vb,
    const int* __restrict__ edge,
    float* __restrict__ out) {
  __shared__ unsigned short kt_s[32 * D_HEAD];   // 4 KB: K tile (bf16)
  __shared__ unsigned short vt_s[32 * D_HEAD];   // 4 KB: V tile (bf16)
  __shared__ int            et_s[32 * 64];       // 8 KB: edge[key][query] tile
  __shared__ unsigned short p_s[4][16 * 32];     // 4 KB: per-wave P re-layout

  const int tid  = threadIdx.x;
  const int lane = tid & 31;
  const int l15  = lane & 15;
  const int wave = tid >> 5;
  const int h    = blockIdx.y;
  const int q0wg = blockIdx.x * 64;
  const int qrow = q0wg + wave * 16 + l15;

  const unsigned short* qh = qb + (size_t)h * N_NODES * D_HEAD;
  const unsigned short* kh = kb + (size_t)h * N_NODES * D_HEAD;
  const unsigned short* vh = vb + (size_t)h * N_NODES * D_HEAD;

  // Q fragments for this wave's 16 rows (features 0..31 and 32..63)
  v16bf aq0, aq1;
#pragma unroll
  for (int j = 0; j < 16; ++j) {
    const int f = amap_k(j, lane);
    aq0[j] = bits2bf(qh[(size_t)qrow * D_HEAD + f]);
    aq1[j] = bits2bf(qh[(size_t)qrow * D_HEAD + 32 + f]);
  }

  float mrow[8], lrow[8];
  v8f o0 = {0.f,0.f,0.f,0.f,0.f,0.f,0.f,0.f};
  v8f o1 = o0, o2 = o0, o3 = o0;
#pragma unroll
  for (int r = 0; r < 8; ++r) { mrow[r] = -3.0e38f; lrow[r] = 0.f; }

  const float scale = 0.125f;  // 1/sqrt(64)
  const unsigned ldsK = lds_off(kt_s);
  const unsigned ldsV = lds_off(vt_s);
  const unsigned ldsE = lds_off(et_s);

  for (int kt = 0; kt < N_NODES / 32; ++kt) {
    const int k0 = kt * 32;

    // ---- TDM: DMA K/V tiles (1D, bf16) + edge sub-tile (2D, strided int32) ----
    if (wave == 0) {
      tdm_load_2d(ldsK, kh + (size_t)k0 * D_HEAD, /*2B*/1, 2048, 0, 2048);
      tdm_load_2d(ldsV, vh + (size_t)k0 * D_HEAD, /*2B*/1, 2048, 0, 2048);
      tdm_load_2d(ldsE, edge + (size_t)k0 * N_NODES + q0wg, /*4B*/2,
                  64, 32, (unsigned long long)N_NODES);
      __builtin_amdgcn_s_wait_tensorcnt(0);
    } else {
      // warm GL2 for the next key tile's edge rows (dominant HBM stream)
      if (k0 + 32 < N_NODES)
        __builtin_prefetch(edge + (size_t)(k0 + 32 + (tid & 31)) * N_NODES + q0wg,
                           0, 1);
    }
    __syncthreads();

    // ---- scores: S[16q x 32k] as two 16x16 WMMA tiles ----
    v8f s0 = {0.f,0.f,0.f,0.f,0.f,0.f,0.f,0.f};
    v8f s1 = s0;
#pragma unroll
    for (int g = 0; g < 2; ++g) {
      v16bf b0, b1;
#pragma unroll
      for (int j = 0; j < 16; ++j) {
        const int krow = g * 16 + l15;                // key within tile (N)
        b0[j] = bits2bf(kt_s[krow * D_HEAD + bmap_k(j, lane)]);
        b1[j] = bits2bf(kt_s[krow * D_HEAD + 32 + bmap_k(j, lane)]);
      }
      v8f sg = {0.f,0.f,0.f,0.f,0.f,0.f,0.f,0.f};
      sg = __builtin_amdgcn_wmma_f32_16x16x32_bf16(false, aq0, false, b0,
                                                   (short)0, sg, false, false);
      sg = __builtin_amdgcn_wmma_f32_16x16x32_bf16(false, aq1, false, b1,
                                                   (short)0, sg, false, false);
      if (g == 0) s0 = sg; else s1 = sg;
    }

    // ---- scale + additive graph mask: edge[key][query]==0 -> -1e4 ----
#pragma unroll
    for (int r = 0; r < 8; ++r) {
      const int m = r + ((lane < 16) ? 0 : 8);            // query row in tile
      const int e0 = et_s[(0 * 16 + l15) * 64 + wave * 16 + m];
      const int e1 = et_s[(1 * 16 + l15) * 64 + wave * 16 + m];
      s0[r] = s0[r] * scale + (e0 ? 0.f : -10000.f);
      s1[r] = s1[r] * scale + (e1 ? 0.f : -10000.f);
    }

    // ---- online softmax (per query row) ----
#pragma unroll
    for (int r = 0; r < 8; ++r) {
      const float mg = fmaxf(half16_max(s0[r]), half16_max(s1[r]));
      const float mnew = fmaxf(mrow[r], mg);
      const float corr = __expf(mrow[r] - mnew);
      const float p0 = __expf(s0[r] - mnew);
      const float p1 = __expf(s1[r] - mnew);
      lrow[r] = lrow[r] * corr + half16_sum(p0) + half16_sum(p1);
      mrow[r] = mnew;
      o0[r] *= corr; o1[r] *= corr; o2[r] *= corr; o3[r] *= corr;
      s0[r] = p0; s1[r] = p1;
    }

    // ---- re-layout P (C/D layout -> A layout) through LDS ----
#pragma unroll
    for (int r = 0; r < 8; ++r) {
      const int m = r + ((lane < 16) ? 0 : 8);
      p_s[wave][m * 32 + l15]      = f2bfbits(s0[r]);
      p_s[wave][m * 32 + 16 + l15] = f2bfbits(s1[r]);
    }
    __syncthreads();

    v16bf ap;
#pragma unroll
    for (int j = 0; j < 16; ++j)
      ap[j] = bits2bf(p_s[wave][l15 * 32 + amap_k(j, lane)]);

    // ---- O += P(16x32) * V(32x64), 4 WMMAs over d-chunks ----
#pragma unroll
    for (int c = 0; c < 4; ++c) {
      v16bf bvf;
#pragma unroll
      for (int j = 0; j < 16; ++j)
        bvf[j] = bits2bf(vt_s[bmap_k(j, lane) * D_HEAD + c * 16 + l15]);
      v8f oc = (c == 0) ? o0 : ((c == 1) ? o1 : ((c == 2) ? o2 : o3));
      oc = __builtin_amdgcn_wmma_f32_16x16x32_bf16(false, ap, false, bvf,
                                                   (short)0, oc, false, false);
      if (c == 0) o0 = oc; else if (c == 1) o1 = oc;
      else if (c == 2) o2 = oc; else o3 = oc;
    }
    __syncthreads();  // protect tiles before next iteration's TDM writes
  }

  // ---- epilogue: normalize and write (N, H*d) fp32 ----
#pragma unroll
  for (int r = 0; r < 8; ++r) {
    const float inv = 1.0f / lrow[r];
    const int m = r + ((lane < 16) ? 0 : 8);
    const int row = q0wg + wave * 16 + m;
    float* op = out + (size_t)row * (N_HEAD * D_HEAD) + h * D_HEAD;
    op[0 * 16 + l15] = o0[r] * inv;
    op[1 * 16 + l15] = o1[r] * inv;
    op[2 * 16 + l15] = o2[r] * inv;
    op[3 * 16 + l15] = o3[r] * inv;
  }
}

// ---------- launcher ----------
extern "C" void kernel_launch(void* const* d_in, const int* in_sizes, int n_in,
                              void* d_out, int out_size, void* d_ws, size_t ws_size,
                              hipStream_t stream) {
  const float* x  = (const float*)d_in[0];
  const int* edge = (const int*)d_in[1];
  const float* Wq = (const float*)d_in[2];
  const float* bq = (const float*)d_in[3];
  const float* Wk = (const float*)d_in[4];
  const float* bk = (const float*)d_in[5];
  const float* Wv = (const float*)d_in[6];
  const float* bv = (const float*)d_in[7];
  float* out = (float*)d_out;

  // bf16 Q/K/V in head-major [h][n][64] layout: 3 * 8192*256 * 2B = 12 MB
  unsigned short* qbuf = (unsigned short*)d_ws;
  unsigned short* kbuf = qbuf + (size_t)N_NODES * D_MODEL;
  unsigned short* vbuf = kbuf + (size_t)N_NODES * D_MODEL;

  qkv_proj_kernel<<<dim3(N_NODES / 16, D_MODEL / 16, 3), 32, 0, stream>>>(
      x, Wq, bq, Wk, bk, Wv, bv, qbuf, kbuf, vbuf);
  gat_attn_kernel<<<dim3(N_NODES / 64, N_HEAD), 128, 0, stream>>>(
      qbuf, kbuf, vbuf, edge, out);
}
